// CausalSelfAttention_38405597561809
// MI455X (gfx1250) — matrix operand order repaired and checked by reading, not compiled, not run on hardware
//
#include <hip/hip_runtime.h>
#include <hip/hip_bf16.h>

typedef __bf16 bf16;
typedef __attribute__((ext_vector_type(16))) __bf16 v16bf;
typedef __attribute__((ext_vector_type(8)))  __bf16 v8bf;
typedef __attribute__((ext_vector_type(8)))  float  v8f;
typedef __attribute__((ext_vector_type(4)))  unsigned int u32x4;
typedef __attribute__((ext_vector_type(4)))  int i32x4;
typedef __attribute__((ext_vector_type(8)))  int i32x8;

static constexpr int Bb = 2, T = 2048, C = 2048, H = 16, HD = 128;
static constexpr int M_ROWS = Bb * T;     // 4096
static constexpr int NQKV   = 3 * C;      // 6144
static constexpr float SCALE = 0.08838834764831845f; // 1/sqrt(128)

__device__ __forceinline__ v16bf cat8(v8bf lo, v8bf hi) {
  v16bf r;
#pragma unroll
  for (int i = 0; i < 8; ++i) { r[i] = lo[i]; r[i + 8] = hi[i]; }
  return r;
}

__device__ __forceinline__ v8f wmma_bf16(v16bf a, v16bf b, v8f c) {
  return __builtin_amdgcn_wmma_f32_16x16x32_bf16(false, a, false, b,
                                                 (short)0, c, false, false);
}

// --------------------------- TDM: 2D tile (tile_d0 x tile_d1) of 2B elements
// gaddr = global address of tile start; stride0/tdim0/tdim1 in elements.
__device__ __forceinline__ void tdm_load_2d(unsigned lds_off, const void* gaddr,
                                            unsigned tile_d0, unsigned tile_d1,
                                            unsigned long long stride0,
                                            unsigned long long tdim0,
                                            unsigned long long tdim1) {
  const unsigned long long ga = (unsigned long long)gaddr;
  u32x4 g0;
  g0[0] = 1u;                                    // count=1, user descriptor
  g0[1] = lds_off;                               // lds_addr (bytes)
  g0[2] = (unsigned)(ga & 0xffffffffu);          // global_addr[31:0]
  g0[3] = (unsigned)((ga >> 32) & 0x1ffffffu)    // global_addr[56:32]
          | (2u << 30);                          // type = 2 ("image")
  i32x8 g1;
  g1[0] = 0x00010000;                            // data_size=1 (2 bytes)
  g1[1] = (int)((tdim0 & 0xffffu) << 16);        // tensor_dim0[15:0]
  g1[2] = (int)(((tdim0 >> 16) & 0xffffu) | ((tdim1 & 0xffffu) << 16));
  g1[3] = (int)(((tdim1 >> 16) & 0xffffu) | ((tile_d0 & 0xffffu) << 16));
  g1[4] = (int)(tile_d1 & 0xffffu);              // tile_dim1 (tile_dim2=0)
  g1[5] = (int)(stride0 & 0xffffffffu);          // tensor_dim0_stride[31:0]
  g1[6] = (int)((stride0 >> 32) & 0xffffu);      // tensor_dim0_stride[47:32]
  g1[7] = 0;
  const i32x4 z4 = {0, 0, 0, 0};
#if __clang_major__ >= 23
  const i32x8 z8 = {0, 0, 0, 0, 0, 0, 0, 0};
  __builtin_amdgcn_tensor_load_to_lds(g0, g1, z4, z4, z8, 0);
#else
  __builtin_amdgcn_tensor_load_to_lds(g0, g1, z4, z4, 0);
#endif
}

// ---------------------------------------------------------------- f32 -> bf16
__global__ void k_cvt(const float* __restrict__ in, bf16* __restrict__ out, int n) {
  int i = blockIdx.x * blockDim.x + threadIdx.x;
  const int stride = gridDim.x * blockDim.x;
  for (; i < n; i += stride) out[i] = (bf16)in[i];
}

// ------------------------------------------------- WMMA GEMM: C = A * W^T
// A: (M x K) bf16 row-major, W: (N x K) bf16 row-major, C: (M x N)
// block = (32,4): 4 waves, tile 64(M) x 64(N); W tile staged by TDM with
// double buffering (tensor DMA overlaps the WMMA stream).
template <bool OUT_BF16>
__global__ __launch_bounds__(128)
void k_gemm(const bf16* __restrict__ A, const bf16* __restrict__ W,
            void* __restrict__ Cout, int N, int K) {
  __shared__ __align__(32) bf16 Btile[2][64][32];
  const int lane = threadIdx.x, wv = threadIdx.y;
  const int hl = lane >> 4, ln = lane & 15;
  const int n0 = blockIdx.x * 64, m0 = blockIdx.y * 64;
  const int mrow = m0 + wv * 16 + ln;

  v8f acc[4] = {};

  if (wv == 0) {  // prologue: DMA first W slab (64 rows x 32 K) into buffer 0
    tdm_load_2d((unsigned)(uintptr_t)&Btile[0][0][0],
                W + (size_t)n0 * K, 32, 64, (unsigned long long)K,
                (unsigned long long)K, (unsigned long long)N);
  }

  for (int k0 = 0, it = 0; k0 < K; k0 += 32, ++it) {
    const int cur = it & 1;
    if (wv == 0) {
      if (k0 + 32 < K) {
        tdm_load_2d((unsigned)(uintptr_t)&Btile[cur ^ 1][0][0],
                    W + (size_t)n0 * K + (k0 + 32), 32, 64,
                    (unsigned long long)K, (unsigned long long)K,
                    (unsigned long long)N);
        __builtin_amdgcn_s_wait_tensorcnt(1);  // current slab landed
      } else {
        __builtin_amdgcn_s_wait_tensorcnt(0);
      }
    }
    __syncthreads();

    // A fragment: row = mrow, K = klo..klo+15
    const int klo = k0 + (hl << 4);
    v8bf alo = *(const v8bf*)(A + (size_t)mrow * K + klo);
    v8bf ahi = *(const v8bf*)(A + (size_t)mrow * K + klo + 8);
    v16bf afrag = cat8(alo, ahi);

#pragma unroll
    for (int nb = 0; nb < 4; ++nb) {
      const v16bf bfrag = *(const v16bf*)&Btile[cur][nb * 16 + ln][hl << 4];
      acc[nb] = wmma_bf16(afrag, bfrag, acc[nb]);
    }
    __syncthreads();
  }

#pragma unroll
  for (int nb = 0; nb < 4; ++nb) {
#pragma unroll
    for (int r = 0; r < 8; ++r) {
      const int row = m0 + wv * 16 + r + (hl << 3);
      const int col = n0 + nb * 16 + ln;
      if (OUT_BF16) ((bf16*)Cout)[(size_t)row * N + col] = (bf16)acc[nb][r];
      else          ((float*)Cout)[(size_t)row * N + col] = acc[nb][r];
    }
  }
}

// ----------------------------------- RMSNorm + RoPE + rearrange (per b,t,h)
// qkv: (B*T, 3C) bf16.  q,k -> (B,H,T,HD) bf16; v -> transposed (B,H,HD,T).
__global__ __launch_bounds__(128)
void k_normrope(const bf16* __restrict__ qkv, const float* __restrict__ qw,
                const float* __restrict__ kw, const float* __restrict__ cosT,
                const float* __restrict__ sinT, bf16* __restrict__ qB,
                bf16* __restrict__ kB, bf16* __restrict__ vT) {
  __shared__ float red[128];
  __shared__ float sh[128];
  const int d = threadIdx.x;
  int bid = blockIdx.x;
  const int h = bid % H; bid /= H;
  const int t = bid % T; bid /= T;
  const int b = bid;

  const size_t rowbase = ((size_t)(b * T + t)) * NQKV + h * HD;
  const float cv = cosT[t * (HD / 2) + (d & 63)];
  const float sv = sinT[t * (HD / 2) + (d & 63)];
  const size_t oidx = ((size_t)((b * H + h) * T) + t) * HD + d;

  for (int which = 0; which < 2; ++which) {
    const float val = (float)qkv[rowbase + which * C + d];
    red[d] = val * val;
    __syncthreads();
    for (int s = 64; s > 0; s >>= 1) {
      if (d < s) red[d] += red[d + s];
      __syncthreads();
    }
    const float nrm = rsqrtf(red[0] * (1.0f / HD) + 1e-6f);
    const float w = (which == 0) ? qw[d] : kw[d];
    const float nv = val * nrm * w;
    sh[d] = nv;
    __syncthreads();
    float o;
    if (d < 64) o = nv * cv - sh[d + 64] * sv;   // x1*c - x2*s
    else        o = nv * cv + sh[d - 64] * sv;   // x2*c + x1*s
    if (which == 0) qB[oidx] = (bf16)o; else kB[oidx] = (bf16)o;
    __syncthreads();
  }
  const float vv = (float)qkv[rowbase + 2 * C + d];
  vT[(size_t)((b * H + h) * HD + d) * T + t] = (bf16)vv;
}

// ---------------------------------------- flash attention, 1 wave = 16 q rows
// grid (B*H, T/64), block (32,4). q,k: (B,H,T,HD); v: (B,H,HD,T); y: (B,T,C).
__global__ __launch_bounds__(128)
void k_attn(const bf16* __restrict__ qB, const bf16* __restrict__ kB,
            const bf16* __restrict__ vT, bf16* __restrict__ yB) {
  __shared__ __align__(32) bf16 P[4][16 * 32];
  const int lane = threadIdx.x;
  const int hl = lane >> 4, ln = lane & 15;
  const int bh = blockIdx.x;
  const int q0 = blockIdx.y * 64 + threadIdx.y * 16;
  bf16* Pl = &P[threadIdx.y][0];

  // resident Q fragments: 4 chunks of K=32
  const bf16* Q = qB + ((size_t)bh * T + q0) * HD;
  v16bf qf[4];
#pragma unroll
  for (int c = 0; c < 4; ++c) {
    const int klo = c * 32 + (hl << 4);
    v8bf lo = *(const v8bf*)(Q + (size_t)ln * HD + klo);
    v8bf hi = *(const v8bf*)(Q + (size_t)ln * HD + klo + 8);
    qf[c] = cat8(lo, hi);
  }

  float mrow[8], lrow[8];
  v8f acc[8] = {};
#pragma unroll
  for (int r = 0; r < 8; ++r) { mrow[r] = -1e30f; lrow[r] = 0.0f; }

  const int ntiles = (q0 + 47) >> 5;        // keys up to q0+15, tiles of 32
  for (int kt = 0; kt < ntiles; ++kt) {
    const int kbase = kt * 32;
    const bf16* Kp = kB + ((size_t)bh * T + kbase) * HD;

    // S = Q K^T for 32 keys (two 16-key WMMA column blocks)
    v8f s0 = {}, s1 = {};
#pragma unroll
    for (int c = 0; c < 4; ++c) {
      const int doff = c * 32 + (hl << 4);
      v16bf b0 = *(const v16bf*)(Kp + (size_t)ln * HD + doff);
      v16bf b1 = *(const v16bf*)(Kp + (size_t)(ln + 16) * HD + doff);
      s0 = wmma_bf16(qf[c], b0, s0);
      s1 = wmma_bf16(qf[c], b1, s1);
    }

    // online softmax (rows live across 16-lane halves; reduce via shfl_xor)
    const int rbase = q0 + (hl << 3);
    const int col0 = kbase + ln;
#pragma unroll
    for (int r = 0; r < 8; ++r) {
      float v0 = s0[r] * SCALE; if (col0 > rbase + r)      v0 = -1e30f;
      float v1 = s1[r] * SCALE; if (col0 + 16 > rbase + r) v1 = -1e30f;
      float mx = fmaxf(v0, v1);
#pragma unroll
      for (int m = 1; m < 16; m <<= 1) mx = fmaxf(mx, __shfl_xor(mx, m, 32));
      const float mnew = fmaxf(mrow[r], mx);
      const float p0 = __expf(v0 - mnew);
      const float p1 = __expf(v1 - mnew);
      float rsum = p0 + p1;
#pragma unroll
      for (int m = 1; m < 16; m <<= 1) rsum += __shfl_xor(rsum, m, 32);
      const float alpha = __expf(mrow[r] - mnew);
      lrow[r] = lrow[r] * alpha + rsum;
      mrow[r] = mnew;
#pragma unroll
      for (int j = 0; j < 8; ++j) acc[j][r] *= alpha;
      // stash P (C-layout) into LDS for re-layout into an A fragment
      Pl[(r + (hl << 3)) * 32 + ln]      = (bf16)p0;
      Pl[(r + (hl << 3)) * 32 + 16 + ln] = (bf16)p1;
    }
    asm volatile("s_wait_dscnt 0" ::: "memory");

    // P as A fragment (16x32, K = 32 keys)
    v8bf plo = *(const v8bf*)(Pl + ln * 32 + (hl << 4));
    v8bf phi = *(const v8bf*)(Pl + ln * 32 + (hl << 4) + 8);
    v16bf pf = cat8(plo, phi);

    // O += P @ V ; V transposed in memory -> contiguous 32B B-fragments
    const bf16* Vp = vT + (size_t)bh * HD * T;
#pragma unroll
    for (int ns = 0; ns < 8; ++ns) {
      const int dn = ns * 16 + ln;
      v16bf bv = *(const v16bf*)(Vp + (size_t)dn * T + kbase + (hl << 4));
      acc[ns] = wmma_bf16(pf, bv, acc[ns]);
    }
  }

  const int h = bh % H, b = bh / H;
#pragma unroll
  for (int ns = 0; ns < 8; ++ns) {
#pragma unroll
    for (int r = 0; r < 8; ++r) {
      const int row = q0 + r + (hl << 3);
      const int dim = ns * 16 + ln;
      yB[((size_t)(b * T + row)) * C + h * HD + dim] = (bf16)(acc[ns][r] / lrow[r]);
    }
  }
}

// ---------------------------------------------------------------------------
extern "C" void kernel_launch(void* const* d_in, const int* in_sizes, int n_in,
                              void* d_out, int out_size, void* d_ws, size_t ws_size,
                              hipStream_t stream) {
  const float* x     = (const float*)d_in[0];
  const float* Wqkv  = (const float*)d_in[1];
  const float* Wproj = (const float*)d_in[2];
  const float* qw    = (const float*)d_in[3];
  const float* kw    = (const float*)d_in[4];
  const float* rc    = (const float*)d_in[5];
  const float* rs    = (const float*)d_in[6];
  float* out = (float*)d_out;

  char* ws = (char*)d_ws;
  size_t off = 0;
  auto alloc = [&](size_t bytes) {
    char* p = ws + off;
    off += (bytes + 255) & ~(size_t)255;
    return p;
  };
  bf16* xb  = (bf16*)alloc((size_t)M_ROWS * C * 2);
  bf16* wqb = (bf16*)alloc((size_t)NQKV * C * 2);
  bf16* wpb = (bf16*)alloc((size_t)C * C * 2);
  bf16* qkv = (bf16*)alloc((size_t)M_ROWS * NQKV * 2);
  bf16* qB  = (bf16*)alloc((size_t)M_ROWS * C * 2);
  bf16* kB  = (bf16*)alloc((size_t)M_ROWS * C * 2);
  bf16* vT  = (bf16*)alloc((size_t)M_ROWS * C * 2);
  bf16* yB  = xb;  // xb is dead after the QKV GEMM; reuse for attention output

  k_cvt<<<2048, 256, 0, stream>>>(x, xb, M_ROWS * C);
  k_cvt<<<2048, 256, 0, stream>>>(Wqkv, wqb, NQKV * C);
  k_cvt<<<2048, 256, 0, stream>>>(Wproj, wpb, C * C);

  k_gemm<true><<<dim3(NQKV / 64, M_ROWS / 64), dim3(32, 4), 0, stream>>>(
      xb, wqb, qkv, NQKV, C);

  k_normrope<<<Bb * T * H, 128, 0, stream>>>(qkv, qw, kw, rc, rs, qB, kB, vT);

  k_attn<<<dim3(Bb * H, T / 64), dim3(32, 4), 0, stream>>>(qB, kB, vT, yB);

  k_gemm<false><<<dim3(C / 64, M_ROWS / 64), dim3(32, 4), 0, stream>>>(
      yB, wpb, out, C, C);
}
